// LSTWithAdapters_88467736363676
// MI455X (gfx1250) — matrix operand order repaired and checked by reading, not compiled, not run on hardware
//
#include <hip/hip_runtime.h>
#include <cstdint>
#include <cstddef>

// ---------------------------------------------------------------------------
// MI455X / gfx1250: wave32, WMMA bf16 16x16x32 fp32-accum + async LDS staging.
// ---------------------------------------------------------------------------

typedef __bf16 bf16;
typedef __attribute__((ext_vector_type(8)))  __bf16 v8bf;
typedef __attribute__((ext_vector_type(16))) __bf16 v16bf;
typedef __attribute__((ext_vector_type(8)))  float  v8f;
typedef __attribute__((vector_size(16))) int i4v;                 // int4
typedef __attribute__((address_space(1))) i4v gl_i4;              // global int4
typedef __attribute__((address_space(3))) i4v ld_i4;              // LDS int4

#define T_DIM 256
#define B_DIM 64
#define H_DIM 1024
#define L_DIM 4
#define DA_DIM 64
#define G4H   (4 * H_DIM)       // 4096
#define MROWS (T_DIM * B_DIM)   // 16384
#define SCAN_NBLK 32

#if defined(__has_builtin)
#  if __has_builtin(__builtin_amdgcn_global_load_async_to_lds_b128) && \
      __has_builtin(__builtin_amdgcn_s_wait_asynccnt)
#    define HAVE_ASYNC_LDS 1
#  endif
#endif
#ifndef HAVE_ASYNC_LDS
#  define HAVE_ASYNC_LDS 0
#  warning "gfx1250 async-to-LDS builtins unavailable; synchronous LDS staging fallback"
#endif

// --------------------------- WMMA fragment helpers -------------------------
// A-matrix 16x32 bf16 (ISA 7.12.2): lane (m = lane%16, half = lane/16) holds
//   A[m][k0 + half*8 .. +7]  and  A[m][k0 + 16 + half*8 .. +7]   (2x16B chunks)
__device__ __forceinline__ v16bf ld_a_frag(const bf16* A, int lda, int m0, int k0) {
  const int lane = threadIdx.x & 31;
  const int m = lane & 15, half = lane >> 4;
  const bf16* p = A + (size_t)(m0 + m) * lda + k0 + half * 8;
  v8bf lo = *reinterpret_cast<const v8bf*>(p);
  v8bf hi = *reinterpret_cast<const v8bf*>(p + 16);
  return __builtin_shufflevector(lo, hi, 0,1,2,3,4,5,6,7,8,9,10,11,12,13,14,15);
}

// B-matrix 32x16 bf16: lane (n = lane%16, half = lane/16) holds column n,
// K = k0 + half*8 .. +7 and k0 + 16 + half*8 .. +7. With B = W^T (W row-major
// [N][K]) these are two 16B chunks of row n of W.
__device__ __forceinline__ v16bf ld_b_frag(const bf16* W, int ldw, int n0, int k0) {
  const int lane = threadIdx.x & 31;
  const int n = lane & 15, half = lane >> 4;
  const bf16* p = W + (size_t)(n0 + n) * ldw + k0 + half * 8;
  v8bf lo = *reinterpret_cast<const v8bf*>(p);
  v8bf hi = *reinterpret_cast<const v8bf*>(p + 16);
  return __builtin_shufflevector(lo, hi, 0,1,2,3,4,5,6,7,8,9,10,11,12,13,14,15);
}

__device__ __forceinline__ v8f wmma_bf16(v16bf a, v16bf b, v8f c) {
  return __builtin_amdgcn_wmma_f32_16x16x32_bf16(false, a, false, b,
                                                 (short)0, c, false, false);
}

__device__ __forceinline__ float sig_f(float x) { return 1.f / (1.f + __expf(-x)); }
__device__ __forceinline__ float tanh_f(float x) {
  x = fminf(10.f, fmaxf(-10.f, x));
  float e = __expf(2.f * x);
  return (e - 1.f) / (e + 1.f);
}

// ------------------------------ utility kernels ----------------------------
__global__ void k_f32_to_bf16(const float* __restrict__ s, bf16* __restrict__ d, int n) {
  for (int i = blockIdx.x * blockDim.x + threadIdx.x; i < n; i += gridDim.x * blockDim.x)
    d[i] = (bf16)s[i];
}

__global__ void k_zero_state(float* __restrict__ h, float* __restrict__ c,
                             bf16* __restrict__ hb, int n) {
  for (int i = blockIdx.x * blockDim.x + threadIdx.x; i < n; i += gridDim.x * blockDim.x) {
    h[i] = 0.f; c[i] = 0.f; hb[i] = (bf16)0.f;
  }
}

__global__ void k_reset_ctr(unsigned* ctr) { *ctr = 0u; }

__global__ void k_gather_out(const float* __restrict__ Xf, const float* __restrict__ h,
                             const float* __restrict__ c, float* __restrict__ out, int n) {
  const int TBH = MROWS * H_DIM, BH = B_DIM * H_DIM;
  for (int i = blockIdx.x * blockDim.x + threadIdx.x; i < n; i += gridDim.x * blockDim.x) {
    float v;
    if (i < TBH)           v = Xf[i];
    else if (i < TBH + BH) v = h[i - TBH];
    else                   v = c[i - TBH - BH];
    out[i] = v;
  }
}

// -------- xg = Xb · Wih^T + (b_ih + b_hh) : [16384,1024] x [1024->4096] ----
// Block: 256 thr = 8 waves as 2(M) x 4(N); wave tile 64x32 (4x2 WMMA tiles).
__global__ __launch_bounds__(256) void k_xg_gemm(
    const bf16* __restrict__ Xb, const bf16* __restrict__ Wih,
    const float* __restrict__ bih, const float* __restrict__ bhh,
    float* __restrict__ XG) {
  const int wave = threadIdx.x >> 5;
  const int m_base = blockIdx.y * 128 + (wave >> 2) * 64;
  const int n_base = blockIdx.x * 128 + (wave & 3) * 32;
  const int lane = threadIdx.x & 31;
  const int cn = lane & 15, ch = lane >> 4;

  v8f acc[4][2];
#pragma unroll
  for (int nt = 0; nt < 2; ++nt) {
    float b = bih[n_base + nt * 16 + cn] + bhh[n_base + nt * 16 + cn];
#pragma unroll
    for (int mt = 0; mt < 4; ++mt)
#pragma unroll
      for (int v = 0; v < 8; ++v) acc[mt][nt][v] = b;
  }
  for (int k0 = 0; k0 < H_DIM; k0 += 32) {
    v16bf a[4], bb[2];
#pragma unroll
    for (int mt = 0; mt < 4; ++mt) a[mt] = ld_a_frag(Xb, H_DIM, m_base + mt * 16, k0);
#pragma unroll
    for (int nt = 0; nt < 2; ++nt) bb[nt] = ld_b_frag(Wih, H_DIM, n_base + nt * 16, k0);
#pragma unroll
    for (int mt = 0; mt < 4; ++mt)
#pragma unroll
      for (int nt = 0; nt < 2; ++nt) acc[mt][nt] = wmma_bf16(a[mt], bb[nt], acc[mt][nt]);
  }
#pragma unroll
  for (int mt = 0; mt < 4; ++mt)
#pragma unroll
    for (int nt = 0; nt < 2; ++nt)
#pragma unroll
      for (int v = 0; v < 8; ++v)
        XG[(size_t)(m_base + mt * 16 + ch * 8 + v) * G4H + n_base + nt * 16 + cn] =
            acc[mt][nt][v];
}

// ----------------------- persistent-grid LSTM scan -------------------------
// Monotonic grid barrier: never resets -> no reset race; counter zeroed per
// launch by k_reset_ctr.
__device__ __forceinline__ void grid_barrier(unsigned* ctr, unsigned target) {
  __syncthreads();
  if (threadIdx.x == 0) {
    __threadfence();
    __hip_atomic_fetch_add(ctr, 1u, __ATOMIC_ACQ_REL, __HIP_MEMORY_SCOPE_AGENT);
    while (__hip_atomic_load(ctr, __ATOMIC_ACQUIRE, __HIP_MEMORY_SCOPE_AGENT) < target)
      __builtin_amdgcn_s_sleep(1);
  }
  __syncthreads();
}

// 32 blocks x 8 waves. Block owns one 32-row half of h (blockIdx&1) staged in
// LDS via async-to-LDS; wave w covers n_base = ((blockIdx>>1)*8 + w)*32.
__global__ __launch_bounds__(256) void k_scan(
    const float* __restrict__ XG, const bf16* __restrict__ Whh,
    float* __restrict__ h, float* __restrict__ c, bf16* __restrict__ hb,
    float* __restrict__ Yf, bf16* __restrict__ Yb,
    float* __restrict__ G, unsigned* __restrict__ ctr) {
  __shared__ bf16 h_lds[32 * H_DIM];              // 64 KB staged h half
  const int wave = threadIdx.x >> 5;
  const int m_base_g = (blockIdx.x & 1) * 32;     // global row base of staged half
  const int n_base = ((blockIdx.x >> 1) * 8 + wave) * 32;
  const int lane = threadIdx.x & 31;
  const int cn = lane & 15, ch = lane >> 4;
  const int tid = blockIdx.x * 256 + threadIdx.x; // 0..8191
  unsigned bar = 0;

  for (int t = 0; t < T_DIM; ++t) {
    // ---- stage h[m_base_g .. +31][:] (64 KB) into LDS (async DMA path) ----
    const bf16* hsrc = hb + (size_t)m_base_g * H_DIM;
#if HAVE_ASYNC_LDS
#pragma unroll
    for (int i = 0; i < 16; ++i) {
      const int off = (threadIdx.x + i * 256) * 8;  // 8 bf16 = 16 B per xfer
      __builtin_amdgcn_global_load_async_to_lds_b128(
          (gl_i4*)(hsrc + off), (ld_i4*)(&h_lds[off]), 0, 0);
    }
    __builtin_amdgcn_s_wait_asynccnt(0);
#else
#pragma unroll
    for (int i = 0; i < 16; ++i) {
      const int off = (threadIdx.x + i * 256) * 8;
      *reinterpret_cast<v8bf*>(&h_lds[off]) = *reinterpret_cast<const v8bf*>(hsrc + off);
    }
#endif
    __syncthreads();

    // ---- Phase A: G = xg_t + h · Whh^T  (one 32x32 tile per wave) ----
    const float* xg_t = XG + (size_t)t * B_DIM * G4H;
    v8f acc[2][2];
#pragma unroll
    for (int mt = 0; mt < 2; ++mt)
#pragma unroll
      for (int nt = 0; nt < 2; ++nt)
#pragma unroll
        for (int v = 0; v < 8; ++v)
          acc[mt][nt][v] =
              xg_t[(size_t)(m_base_g + mt * 16 + ch * 8 + v) * G4H + n_base + nt * 16 + cn];
    for (int k0 = 0; k0 < H_DIM; k0 += 32) {
      v16bf a0 = ld_a_frag(h_lds, H_DIM, 0, k0);    // ds_load path
      v16bf a1 = ld_a_frag(h_lds, H_DIM, 16, k0);
      v16bf b0 = ld_b_frag(Whh, H_DIM, n_base, k0); // L2-resident weights
      v16bf b1 = ld_b_frag(Whh, H_DIM, n_base + 16, k0);
      acc[0][0] = wmma_bf16(a0, b0, acc[0][0]);
      acc[0][1] = wmma_bf16(a0, b1, acc[0][1]);
      acc[1][0] = wmma_bf16(a1, b0, acc[1][0]);
      acc[1][1] = wmma_bf16(a1, b1, acc[1][1]);
    }
#pragma unroll
    for (int mt = 0; mt < 2; ++mt)
#pragma unroll
      for (int nt = 0; nt < 2; ++nt)
#pragma unroll
        for (int v = 0; v < 8; ++v)
          G[(size_t)(m_base_g + mt * 16 + ch * 8 + v) * G4H + n_base + nt * 16 + cn] =
              acc[mt][nt][v];

    bar += SCAN_NBLK; grid_barrier(ctr, bar);

    // ---- Phase B: gates + state update, 8 elements of [B*H]=65536/thread ----
    float* yf_t = Yf + (size_t)t * B_DIM * H_DIM;
    bf16*  yb_t = Yb + (size_t)t * B_DIM * H_DIM;
#pragma unroll
    for (int i = 0; i < 8; ++i) {
      int e = tid + i * 8192;
      int bb = e >> 10, j = e & (H_DIM - 1);
      const float* gr = G + (size_t)bb * G4H;
      float gi = gr[j], gf = gr[H_DIM + j], gg = gr[2 * H_DIM + j], go = gr[3 * H_DIM + j];
      float c_new = sig_f(gf) * c[e] + sig_f(gi) * tanh_f(gg);
      float h_new = sig_f(go) * tanh_f(c_new);
      c[e] = c_new; h[e] = h_new;
      bf16 hnb = (bf16)h_new;
      hb[e] = hnb; yf_t[e] = h_new; yb_t[e] = hnb;
    }
    bar += SCAN_NBLK; grid_barrier(ctr, bar);
  }
}

// ---------- adapter down: D = relu(Y·Wd^T + bd) -> bf16 [16384,64] ---------
__global__ __launch_bounds__(256) void k_down(
    const bf16* __restrict__ Yb, const bf16* __restrict__ Wd,
    const float* __restrict__ bd, bf16* __restrict__ Db) {
  const int wave = threadIdx.x >> 5;
  const int m_base = blockIdx.x * 128 + (wave >> 1) * 32;
  const int n_base = (wave & 1) * 32;
  const int lane = threadIdx.x & 31;
  const int cn = lane & 15, ch = lane >> 4;

  v8f acc[2][2];
#pragma unroll
  for (int nt = 0; nt < 2; ++nt) {
    float b = bd[n_base + nt * 16 + cn];
#pragma unroll
    for (int mt = 0; mt < 2; ++mt)
#pragma unroll
      for (int v = 0; v < 8; ++v) acc[mt][nt][v] = b;
  }
  for (int k0 = 0; k0 < H_DIM; k0 += 32) {
    v16bf a0 = ld_a_frag(Yb, H_DIM, m_base, k0);
    v16bf a1 = ld_a_frag(Yb, H_DIM, m_base + 16, k0);
    v16bf b0 = ld_b_frag(Wd, H_DIM, n_base, k0);
    v16bf b1 = ld_b_frag(Wd, H_DIM, n_base + 16, k0);
    acc[0][0] = wmma_bf16(a0, b0, acc[0][0]);
    acc[0][1] = wmma_bf16(a0, b1, acc[0][1]);
    acc[1][0] = wmma_bf16(a1, b0, acc[1][0]);
    acc[1][1] = wmma_bf16(a1, b1, acc[1][1]);
  }
#pragma unroll
  for (int mt = 0; mt < 2; ++mt)
#pragma unroll
    for (int nt = 0; nt < 2; ++nt)
#pragma unroll
      for (int v = 0; v < 8; ++v) {
        float r = fmaxf(acc[mt][nt][v], 0.f);
        Db[(size_t)(m_base + mt * 16 + ch * 8 + v) * DA_DIM + n_base + nt * 16 + cn] = (bf16)r;
      }
}

// ------- adapter up + residual: Z = Y + D·Wu^T + bu  (K = 64) --------------
__global__ __launch_bounds__(256) void k_up(
    const bf16* __restrict__ Db, const bf16* __restrict__ Wu,
    const float* __restrict__ bu, const float* __restrict__ Yf,
    float* __restrict__ Z) {
  const int wave = threadIdx.x >> 5;
  const int m_base = blockIdx.y * 64 + (wave >> 2) * 32;
  const int n_base = blockIdx.x * 128 + (wave & 3) * 32;
  const int lane = threadIdx.x & 31;
  const int cn = lane & 15, ch = lane >> 4;

  v8f acc[2][2];
#pragma unroll
  for (int nt = 0; nt < 2; ++nt) {
    float b = bu[n_base + nt * 16 + cn];
#pragma unroll
    for (int mt = 0; mt < 2; ++mt)
#pragma unroll
      for (int v = 0; v < 8; ++v) acc[mt][nt][v] = b;
  }
#pragma unroll
  for (int k0 = 0; k0 < DA_DIM; k0 += 32) {
    v16bf a0 = ld_a_frag(Db, DA_DIM, m_base, k0);
    v16bf a1 = ld_a_frag(Db, DA_DIM, m_base + 16, k0);
    v16bf b0 = ld_b_frag(Wu, DA_DIM, n_base, k0);
    v16bf b1 = ld_b_frag(Wu, DA_DIM, n_base + 16, k0);
    acc[0][0] = wmma_bf16(a0, b0, acc[0][0]);
    acc[0][1] = wmma_bf16(a0, b1, acc[0][1]);
    acc[1][0] = wmma_bf16(a1, b0, acc[1][0]);
    acc[1][1] = wmma_bf16(a1, b1, acc[1][1]);
  }
#pragma unroll
  for (int mt = 0; mt < 2; ++mt)
#pragma unroll
    for (int nt = 0; nt < 2; ++nt)
#pragma unroll
      for (int v = 0; v < 8; ++v) {
        size_t idx = (size_t)(m_base + mt * 16 + ch * 8 + v) * H_DIM + n_base + nt * 16 + cn;
        Z[idx] = acc[mt][nt][v] + Yf[idx];
      }
}

// ------------- layernorm over H=1024 per row; emits fp32 + bf16 ------------
__global__ __launch_bounds__(256) void k_ln(
    const float* __restrict__ Z, const float* __restrict__ gamma,
    const float* __restrict__ beta, float* __restrict__ Xf, bf16* __restrict__ Xb) {
  const int row = blockIdx.x;
  const float* z = Z + (size_t)row * H_DIM;
  __shared__ float ssum[256], ssq[256];
  float vals[4], s = 0.f, s2 = 0.f;
#pragma unroll
  for (int i = 0; i < 4; ++i) {
    float v = z[threadIdx.x + i * 256];
    vals[i] = v; s += v; s2 += v * v;
  }
  ssum[threadIdx.x] = s; ssq[threadIdx.x] = s2;
  __syncthreads();
  for (int off = 128; off > 0; off >>= 1) {
    if ((int)threadIdx.x < off) {
      ssum[threadIdx.x] += ssum[threadIdx.x + off];
      ssq[threadIdx.x]  += ssq[threadIdx.x + off];
    }
    __syncthreads();
  }
  float mu = ssum[0] * (1.f / H_DIM);
  float var = ssq[0] * (1.f / H_DIM) - mu * mu;
  float inv = rsqrtf(var + 1e-5f);
#pragma unroll
  for (int i = 0; i < 4; ++i) {
    int col = threadIdx.x + i * 256;
    float nv = (vals[i] - mu) * inv * gamma[col] + beta[col];
    Xf[(size_t)row * H_DIM + col] = nv;
    Xb[(size_t)row * H_DIM + col] = (bf16)nv;
  }
}

// --------------------------------- host ------------------------------------
extern "C" void kernel_launch(void* const* d_in, const int* in_sizes, int n_in,
                              void* d_out, int out_size, void* d_ws, size_t ws_size,
                              hipStream_t stream) {
  const float* x     = (const float*)d_in[0];
  const float* W_ih  = (const float*)d_in[1];
  const float* W_hh  = (const float*)d_in[2];
  const float* b_ih  = (const float*)d_in[3];
  const float* b_hh  = (const float*)d_in[4];
  const float* Wd    = (const float*)d_in[5];
  const float* bd    = (const float*)d_in[6];
  const float* Wu    = (const float*)d_in[7];
  const float* bu    = (const float*)d_in[8];
  const float* gamma = (const float*)d_in[9];
  const float* beta  = (const float*)d_in[10];
  float* out = (float*)d_out;

  char* p = (char*)d_ws;
  auto carve = [&](size_t bytes) -> char* {
    char* r = p; p += (bytes + 255) & ~(size_t)255; return r;
  };
  bf16* Wih_b = (bf16*)carve((size_t)L_DIM * G4H * H_DIM * 2);
  bf16* Whh_b = (bf16*)carve((size_t)L_DIM * G4H * H_DIM * 2);
  bf16* Wd_b  = (bf16*)carve((size_t)L_DIM * DA_DIM * H_DIM * 2);
  bf16* Wu_b  = (bf16*)carve((size_t)L_DIM * H_DIM * DA_DIM * 2);
  bf16* xb    = (bf16*)carve((size_t)MROWS * H_DIM * 2);
  bf16* yb    = (bf16*)carve((size_t)MROWS * H_DIM * 2);
  bf16* db    = (bf16*)carve((size_t)MROWS * DA_DIM * 2);
  bf16* hb    = (bf16*)carve((size_t)B_DIM * H_DIM * 2);
  float* xf   = (float*)carve((size_t)MROWS * H_DIM * 4);
  float* yf   = (float*)carve((size_t)MROWS * H_DIM * 4);
  float* zf   = (float*)carve((size_t)MROWS * H_DIM * 4);
  float* xg   = (float*)carve((size_t)MROWS * G4H * 4);
  float* G    = (float*)carve((size_t)B_DIM * G4H * 4);
  float* h    = (float*)carve((size_t)B_DIM * H_DIM * 4);
  float* c    = (float*)carve((size_t)B_DIM * H_DIM * 4);
  unsigned* ctr = (unsigned*)carve(256);

  auto cvt = [&](const float* s, bf16* d, int n) {
    int blocks = (n + 255) / 256; if (blocks > 16384) blocks = 16384;
    k_f32_to_bf16<<<blocks, 256, 0, stream>>>(s, d, n);
  };
  cvt(W_ih, Wih_b, L_DIM * G4H * H_DIM);
  cvt(W_hh, Whh_b, L_DIM * G4H * H_DIM);
  cvt(Wd,   Wd_b,  L_DIM * DA_DIM * H_DIM);
  cvt(Wu,   Wu_b,  L_DIM * H_DIM * DA_DIM);
  cvt(x,    xb,    MROWS * H_DIM);
  k_zero_state<<<256, 256, 0, stream>>>(h, c, hb, B_DIM * H_DIM);

  for (int l = 0; l < L_DIM; ++l) {
    const bf16* Wih_l = Wih_b + (size_t)l * G4H * H_DIM;
    const bf16* Whh_l = Whh_b + (size_t)l * G4H * H_DIM;
    const bf16* Wd_l  = Wd_b  + (size_t)l * DA_DIM * H_DIM;
    const bf16* Wu_l  = Wu_b  + (size_t)l * H_DIM * DA_DIM;
    const float* bih_l = b_ih + (size_t)l * G4H;
    const float* bhh_l = b_hh + (size_t)l * G4H;
    const float* bd_l  = bd + (size_t)l * DA_DIM;
    const float* bu_l  = bu + (size_t)l * H_DIM;
    const float* g_l   = gamma + (size_t)l * H_DIM;
    const float* be_l  = beta  + (size_t)l * H_DIM;

    k_xg_gemm<<<dim3(G4H / 128, MROWS / 128), 256, 0, stream>>>(xb, Wih_l, bih_l, bhh_l, xg);
    k_reset_ctr<<<1, 1, 0, stream>>>(ctr);
    k_scan<<<SCAN_NBLK, 256, 0, stream>>>(xg, Whh_l, h, c, hb, yf, yb, G, ctr);
    k_down<<<MROWS / 128, 256, 0, stream>>>(yb, Wd_l, bd_l, db);
    k_up<<<dim3(H_DIM / 128, MROWS / 64), 256, 0, stream>>>(db, Wu_l, bu_l, yf, zf);
    k_ln<<<MROWS, 256, 0, stream>>>(zf, g_l, be_l, xf, xb);
  }

  const int n_out = MROWS * H_DIM + 2 * B_DIM * H_DIM;
  int blocks = (n_out + 255) / 256; if (blocks > 32768) blocks = 32768;
  k_gather_out<<<blocks, 256, 0, stream>>>(xf, h, c, out, n_out);
}
